// Attention_14809047236844
// MI455X (gfx1250) — matrix operand order repaired and checked by reading, not compiled, not run on hardware
//
#include <hip/hip_runtime.h>
#include <math.h>

#define EMBED 128
#define NLOC  360
#define NB    4
#define TT    128
#define FF    512

typedef __bf16 bf16;
typedef bf16  v16bf __attribute__((ext_vector_type(16)));
typedef float v8f   __attribute__((ext_vector_type(8)));

// ---------------------------------------------------------------------------
// CDNA5 async global->LDS copy (GLOBAL_LOAD_ASYNC_TO_LDS_B128, ASYNCcnt).
// LDS dest must be 16B-aligned; low 32 bits of a generic shared pointer are
// the wave-relative LDS byte address (ISA 10.2 aperture truncation).
// ---------------------------------------------------------------------------
__device__ __forceinline__ void async_b128(void* lds, const void* g) {
  unsigned l = (unsigned)(unsigned long long)lds;
  asm volatile("global_load_async_to_lds_b128 %0, %1, off"
               :: "v"(l), "v"(g) : "memory");
}
__device__ __forceinline__ void wait_async0() {
  asm volatile("s_wait_asynccnt 0" ::: "memory");
}

// ---------------------------------------------------------------------------
// WMMA fragment loaders (wave32; contiguous k order chosen here, same for A and B)
// A (16x32 bf16): lane L holds row M=L%16; lanes 0-15 hold K=0..15,
// lanes 16-31 hold K=16..31 (2 per VGPR).
// B (32x16 bf16): lane L holds col N=L%16; lanes 0-15 K=0..15, 16-31 K=16..31.
// C/D (16x16 f32): VGPR r holds M=(lane/16)*8+r, N=lane%16.
// ---------------------------------------------------------------------------
__device__ __forceinline__ v16bf load_a_rm(const bf16* src, int ld, int lane) {
  int mrow = lane & 15, half = lane >> 4;
  const bf16* row = src + mrow * ld;
  v16bf a;
#pragma unroll
  for (int i = 0; i < 16; ++i) {
    int k = (half << 4) + i;
    a[i] = row[k];
  }
  return a;
}

__device__ __forceinline__ v16bf load_a_kt(const bf16* src, int ld, int lane) {
  // A[m][k], source stored k-major [k][m]
  int mrow = lane & 15, half = lane >> 4;
  v16bf a;
#pragma unroll
  for (int i = 0; i < 16; ++i) {
    int k = (half << 4) + i;
    a[i] = src[k * ld + mrow];
  }
  return a;
}

__device__ __forceinline__ v16bf load_b_km(const bf16* src, int ld, int lane) {
  // B[k][n], source stored k-major [k][n]
  int ncol = lane & 15, half = lane >> 4;
  v16bf b;
#pragma unroll
  for (int i = 0; i < 16; ++i) {
    int k = (half << 4) + i;
    b[i] = src[k * ld + ncol];
  }
  return b;
}

__device__ __forceinline__ v16bf load_b_nm(const bf16* src, int ld, int lane) {
  // B[k][n], source stored n-major [n][k] (rows contiguous in k -> b128 loads)
  int ncol = lane & 15, half = lane >> 4;
  const bf16* row = src + ncol * ld + (half << 4);
  v16bf b;
#pragma unroll
  for (int i = 0; i < 16; ++i) b[i] = row[i];
  return b;
}

__device__ __forceinline__ v8f wmma_bf16(v16bf a, v16bf b, v8f c) {
  return __builtin_amdgcn_wmma_f32_16x16x32_bf16(false, a, false, b,
                                                 (short)0, c, false, false);
}

// ---------------------------------------------------------------------------
// Kernel 1: projection P[n][e][t][f] = sum_e' X[n,t,f,e'] * W[e][e']  (bf16 out)
// Block: 64 consecutive (n,t,f) rows x all 128 output channels.
// (f32->bf16 conversion happens on the way into LDS, so VGPR staging here.)
// ---------------------------------------------------------------------------
__global__ __launch_bounds__(256) void k_proj(const float* __restrict__ X,
                                              const float* __restrict__ W,
                                              bf16* __restrict__ P) {
  __shared__ bf16 Xt[64][132];
  __shared__ bf16 WtOt[128 * 132];   // Wt during GEMM, reused as transposed C stage
  int tid = threadIdx.x, wave = tid >> 5, lane = tid & 31;
  long m0 = (long)blockIdx.x * 64;

  const float* Xb = X + m0 * EMBED;
  for (int i = tid; i < 64 * 128; i += 256)
    Xt[i >> 7][i & 127] = (bf16)Xb[i];
  bf16 (*Wt)[132] = (bf16(*)[132])WtOt;
  for (int i = tid; i < 128 * 128; i += 256)
    Wt[i >> 7][i & 127] = (bf16)W[i];
  __syncthreads();

  int mi = wave & 3, nbase = (wave >> 2) * 4;
  v8f acc[4];
#pragma unroll
  for (int j = 0; j < 4; ++j) {
    int ni = nbase + j;
    v8f c = {};
#pragma unroll
    for (int kk = 0; kk < 4; ++kk) {
      v16bf a = load_a_rm(&Xt[mi * 16][kk * 32], 132, lane);
      v16bf b = load_b_nm(&Wt[ni * 16][kk * 32], 132, lane);
      c = wmma_bf16(a, b, c);
    }
    acc[j] = c;
  }
  __syncthreads();   // all waves done reading Wt before aliasing it

  bf16 (*Ot)[68] = (bf16(*)[68])WtOt;  // Ot[e][m], 128x68 fits in Wt storage
  int half = lane >> 4;
#pragma unroll
  for (int j = 0; j < 4; ++j) {
    int ncol = (nbase + j) * 16 + (lane & 15);
#pragma unroll
    for (int r = 0; r < 8; ++r)
      Ot[ncol][mi * 16 + half * 8 + r] = (bf16)acc[j][r];
  }
  __syncthreads();

  long ntq = m0 / FF; int f0 = (int)(m0 % FF);
  int n = (int)(ntq / TT), t = (int)(ntq % TT);
  bf16* Pb = P + ((long)n * EMBED) * TT * FF + (long)t * FF + f0;
  for (int i = tid; i < 128 * 64; i += 256) {
    int e = i >> 6, m = i & 63;
    Pb[(long)e * TT * FF + m] = Ot[e][m];  // coalesced: contiguous in f
  }
}

// ---------------------------------------------------------------------------
// Kernel 2: energy[n,e,q,k] = sum_t Qp[n,e,t,q] * Kp[n,e,t,k]   (f32 out)
// Block: (n, e, q0 tile of 64) x full k=512 (streamed in 128-wide tiles).
// Tiles are moved with async global->LDS (no VGPR staging).
// ---------------------------------------------------------------------------
__global__ __launch_bounds__(256) void k_energy(const bf16* __restrict__ Qp,
                                                const bf16* __restrict__ Kp,
                                                float* __restrict__ Eng) {
  __shared__ bf16 Qt[128][72];    // [t][q]   (144B rows, 16B aligned)
  __shared__ bf16 Kt[128][136];   // [t][k]   (272B rows, 16B aligned)
  int tid = threadIdx.x, wave = tid >> 5, lane = tid & 31;
  int bid = blockIdx.x;
  int q0 = (bid & 7) * 64;
  int e  = (bid >> 3) & (EMBED - 1);
  int n  = bid >> 10;

  const bf16* Qb = Qp + ((long)(n * EMBED + e)) * TT * FF;
  const bf16* Kb = Kp + ((long)(n * EMBED + e)) * TT * FF;
  // async-copy Q tile: 128 rows x 64 bf16 = 1024 x 16B chunks
  for (int i = tid; i < 1024; i += 256) {
    int t = i >> 3, c = (i & 7) * 8;
    async_b128(&Qt[t][c], Qb + (long)t * FF + q0 + c);
  }

  int mi = wave & 3, nbase = (wave >> 2) * 4;
  int half = lane >> 4;
  float* Ebase = Eng + (((long)(n * EMBED + e)) * FF + q0) * FF;

  for (int k0 = 0; k0 < FF; k0 += 128) {
    __syncthreads();   // previous Kt consumers done before overwrite
    // async-copy K tile: 128 rows x 128 bf16 = 2048 x 16B chunks
    for (int i = tid; i < 2048; i += 256) {
      int t = i >> 4, c = (i & 15) * 8;
      async_b128(&Kt[t][c], Kb + (long)t * FF + k0 + c);
    }
    wait_async0();     // this wave's async copies (incl. Qt on iter 0) landed
    __syncthreads();   // all waves' copies visible
#pragma unroll
    for (int j = 0; j < 4; ++j) {
      int ni = nbase + j;
      v8f c = {};
#pragma unroll
      for (int kk = 0; kk < 4; ++kk) {   // contract t = 128
        v16bf a = load_a_kt(&Qt[kk * 32][mi * 16], 72, lane);
        v16bf b = load_b_km(&Kt[kk * 32][ni * 16], 136, lane);
        c = wmma_bf16(a, b, c);
      }
      int kcol = k0 + ni * 16 + (lane & 15);
#pragma unroll
      for (int r = 0; r < 8; ++r) {
        int q = mi * 16 + half * 8 + r;
        Ebase[(long)q * FF + kcol] = c[r];   // 16-lane x 4B contiguous runs
      }
    }
  }
}

// ---------------------------------------------------------------------------
// Kernel 3: softmax over k (512) per row; one wave32 per row; bf16 out.
// ---------------------------------------------------------------------------
__global__ __launch_bounds__(256) void k_softmax(const float* __restrict__ Eng,
                                                 bf16* __restrict__ Att) {
  int wave = threadIdx.x >> 5, lane = threadIdx.x & 31;
  long row = (long)blockIdx.x * 8 + wave;
  const float* er = Eng + row * FF + lane * 16;  // wave reads contiguous 2KB
  float v[16];
  float mx = -3.0e38f;
#pragma unroll
  for (int i = 0; i < 16; ++i) { v[i] = er[i]; mx = fmaxf(mx, v[i]); }
#pragma unroll
  for (int off = 16; off >= 1; off >>= 1) mx = fmaxf(mx, __shfl_xor(mx, off, 32));
  const float is = 0.08838834764831845f;   // 1/sqrt(EMBED)
  float s = 0.f;
#pragma unroll
  for (int i = 0; i < 16; ++i) { v[i] = __expf((v[i] - mx) * is); s += v[i]; }
#pragma unroll
  for (int off = 16; off >= 1; off >>= 1) s += __shfl_xor(s, off, 32);
  float inv = 1.0f / s;
  bf16* ar = Att + row * FF + lane * 16;
#pragma unroll
  for (int i = 0; i < 16; ++i) ar[i] = (bf16)(v[i] * inv);
}

// ---------------------------------------------------------------------------
// Kernel 4: Out[n,e,t,q] = sum_k Vp[n,e,t,k] * Att[n,e,q,k]   (bf16 out)
// Block: (n, e, q0 tile of 128) x full t=128; k streamed in 64-wide tiles.
// Async global->LDS tile movement.
// ---------------------------------------------------------------------------
__global__ __launch_bounds__(256) void k_av(const bf16* __restrict__ Att,
                                            const bf16* __restrict__ Vp,
                                            bf16* __restrict__ Out) {
  __shared__ bf16 Vt[128][72];   // [t][k]
  __shared__ bf16 At[128][72];   // [q][k]
  int tid = threadIdx.x, wave = tid >> 5, lane = tid & 31;
  int bid = blockIdx.x;
  int q0 = (bid & 3) * 128;
  int e  = (bid >> 2) & (EMBED - 1);
  int n  = bid >> 9;

  const bf16* Vb = Vp + ((long)(n * EMBED + e)) * TT * FF;
  const bf16* Ab = Att + (((long)(n * EMBED + e)) * FF + q0) * FF;

  v8f c[8];
  v8f zero = {};
#pragma unroll
  for (int j = 0; j < 8; ++j) c[j] = zero;

  for (int k0 = 0; k0 < FF; k0 += 64) {
    __syncthreads();
    for (int i = tid; i < 1024; i += 256) {       // V tile: 128 x 64 bf16
      int t = i >> 3, c2 = (i & 7) * 8;
      async_b128(&Vt[t][c2], Vb + (long)t * FF + k0 + c2);
    }
    for (int i = tid; i < 1024; i += 256) {       // A tile: 128 x 64 bf16
      int q = i >> 3, c2 = (i & 7) * 8;
      async_b128(&At[q][c2], Ab + (long)q * FF + k0 + c2);
    }
    wait_async0();
    __syncthreads();
#pragma unroll
    for (int ni = 0; ni < 8; ++ni) {
#pragma unroll
      for (int kk = 0; kk < 2; ++kk) {
        v16bf a = load_a_rm(&Vt[wave * 16][kk * 32], 72, lane);
        v16bf b = load_b_nm(&At[ni * 16][kk * 32], 72, lane);
        c[ni] = wmma_bf16(a, b, c[ni]);
      }
    }
  }

  int half = lane >> 4;
  bf16* Ob = Out + ((long)(n * EMBED + e)) * TT * FF;
#pragma unroll
  for (int ni = 0; ni < 8; ++ni) {
    int q = q0 + ni * 16 + (lane & 15);
#pragma unroll
    for (int r = 0; r < 8; ++r) {
      int t = wave * 16 + half * 8 + r;
      Ob[(long)t * FF + q] = (bf16)c[ni][r];
    }
  }
}

// ---------------------------------------------------------------------------
// Kernel 5: Y[n,t,q,l] = sum_e Out[n,e,t,q] * Wo[l,e] + bo[l]   (f32 out)
// Block: 128 consecutive (n,t,q) rows x all 360 (padded to 368) cols.
// A tile async-copied to LDS; Wo fragments streamed from global (L2-resident).
// ---------------------------------------------------------------------------
__device__ __forceinline__ v16bf load_b_wo(const float* __restrict__ Wo,
                                           int n0, int k0, int lane) {
  int l = n0 + (lane & 15);
  int half = lane >> 4;
  v16bf b;
  if (l < NLOC) {
    const float* row = Wo + (long)l * EMBED + k0 + half * 16;
#pragma unroll
    for (int i = 0; i < 16; ++i) b[i] = (bf16)row[i];
  } else {
#pragma unroll
    for (int i = 0; i < 16; ++i) b[i] = (bf16)0.f;
  }
  return b;
}

__global__ __launch_bounds__(256) void k_out(const bf16* __restrict__ Out,
                                             const float* __restrict__ Wo,
                                             const float* __restrict__ bo,
                                             float* __restrict__ Y) {
  __shared__ bf16 Ot[128][136];   // [e][q] -> k-major for the A fragment
  int tid = threadIdx.x, wave = tid >> 5, lane = tid & 31;
  long m0 = (long)blockIdx.x * 128;
  long ntq = m0 / FF; int q0 = (int)(m0 % FF);
  int n = (int)(ntq / TT), t = (int)(ntq % TT);

  // warm L2 with the output weights (global_prefetch_b8)
  __builtin_prefetch(Wo + (long)tid * 180, 0, 3);

  const bf16* Ob = Out + ((long)n * EMBED) * TT * FF + (long)t * FF + q0;
  for (int i = tid; i < 2048; i += 256) {   // 128 x 128 bf16 = 2048 x 16B
    int e = i >> 4, c = (i & 15) * 8;
    async_b128(&Ot[e][c], Ob + (long)e * TT * FF + c);
  }
  wait_async0();
  __syncthreads();

  float* Yb = Y + ((long)(n * TT + t) * FF + q0) * NLOC;
  int half = lane >> 4;
  for (int ni = wave; ni < 23; ni += 8) {   // 23 n-tiles cover 368 >= 360
    int l = ni * 16 + (lane & 15);
#pragma unroll
    for (int mi = 0; mi < 8; ++mi) {
      v8f c = {};
#pragma unroll
      for (int kk = 0; kk < 4; ++kk) {
        v16bf a = load_a_kt(&Ot[kk * 32][mi * 16], 136, lane);
        v16bf b = load_b_wo(Wo, ni * 16, kk * 32, lane);
        c = wmma_bf16(a, b, c);
      }
      if (l < NLOC) {
        float bias = bo[l];
#pragma unroll
        for (int r = 0; r < 8; ++r) {
          int q = mi * 16 + half * 8 + r;
          Yb[(long)q * NLOC + l] = c[r] + bias;
        }
      }
    }
  }
}

// ---------------------------------------------------------------------------
// Host launcher
// d_in order: value, key, query, Wv, Wk, Wq, Wo, bo
// Workspace layout:
//   [0,64M)    Qp bf16 [N][E][T][F]
//   [64,128M)  Kp
//   [128,192M) Vp
//   [192,704M) Eng f32 [N][E][F][F]   (reused as Out bf16 after softmax)
//   [704,960M) Att bf16 [N][E][F][F]
// ---------------------------------------------------------------------------
extern "C" void kernel_launch(void* const* d_in, const int* in_sizes, int n_in,
                              void* d_out, int out_size, void* d_ws, size_t ws_size,
                              hipStream_t stream) {
  const float* value = (const float*)d_in[0];
  const float* key_  = (const float*)d_in[1];
  const float* query = (const float*)d_in[2];
  const float* Wv    = (const float*)d_in[3];
  const float* Wk    = (const float*)d_in[4];
  const float* Wq    = (const float*)d_in[5];
  const float* Wo    = (const float*)d_in[6];
  const float* bo    = (const float*)d_in[7];
  float* Y = (float*)d_out;

  char* ws = (char*)d_ws;
  const size_t SZ_P = (size_t)NB * EMBED * TT * FF * sizeof(bf16);   // 64 MB
  const size_t SZ_E = (size_t)NB * EMBED * FF * FF * sizeof(float);  // 512 MB
  bf16*  Qp  = (bf16*)(ws);
  bf16*  Kp  = (bf16*)(ws + SZ_P);
  bf16*  Vp  = (bf16*)(ws + 2 * SZ_P);
  float* Eng = (float*)(ws + 3 * SZ_P);
  bf16*  Att = (bf16*)(ws + 3 * SZ_P + SZ_E);
  bf16*  Out = (bf16*)Eng;   // Eng dead after softmax; reuse for Out

  dim3 blk(256);
  k_proj<<<dim3((NB * TT * FF) / 64), blk, 0, stream>>>(query, Wq, Qp);
  k_proj<<<dim3((NB * TT * FF) / 64), blk, 0, stream>>>(key_,  Wk, Kp);
  k_proj<<<dim3((NB * TT * FF) / 64), blk, 0, stream>>>(value, Wv, Vp);
  k_energy<<<dim3(NB * EMBED * (FF / 64)), blk, 0, stream>>>(Qp, Kp, Eng);
  k_softmax<<<dim3((NB * EMBED * FF) / 8), blk, 0, stream>>>(Eng, Att);
  k_av<<<dim3(NB * EMBED * (FF / 128)), blk, 0, stream>>>(Att, Vp, Out);
  k_out<<<dim3((NB * TT * FF) / 128), blk, 0, stream>>>(Out, Wo, bo, Y);
}
